// CGGRUNet_43130061586840
// MI455X (gfx1250) — matrix operand order repaired and checked by reading, not compiled, not run on hardware
//
#include <hip/hip_runtime.h>
#include <hip/hip_bf16.h>
#include <cfloat>
#include <cstdint>

// ---- problem constants (from reference) ----
#define ND    10000     // nodes
#define NE    64000     // edges
#define FN    92        // node feature len
#define FE    41        // edge feature len
#define DD    64        // D
#define HIDC  128       // HID
#define WEN   4096      // D*D
#define NSEGC 128       // NSEG
#define MT    (NE/16)   // 4000 M-tiles of GEMM
#define NT    (WEN/16)  // 256 N-tiles

typedef __attribute__((ext_vector_type(16))) _Float16 v16h;
typedef __attribute__((ext_vector_type(8)))  float    v8f;

__device__ __forceinline__ float sigf(float x){ return 1.0f/(1.0f+expf(-x)); }

// order-preserving float->uint encoding for atomic max over signed floats
__device__ __forceinline__ unsigned enc_f(float f){
  unsigned b = __float_as_uint(f);
  return (b & 0x80000000u) ? ~b : (b | 0x80000000u);
}
__device__ __forceinline__ float dec_f(unsigned k){
  unsigned b = (k & 0x80000000u) ? (k & 0x7FFFFFFFu) : ~k;
  return __uint_as_float(b);
}

// ---- 1) out = h = relu(x @ lin0_w.T + lin0_b) ----
__global__ void k_lin0(const float* __restrict__ x, const float* __restrict__ w,
                       const float* __restrict__ b, float* __restrict__ out,
                       float* __restrict__ h){
  int tid = blockIdx.x*blockDim.x + threadIdx.x;
  if (tid >= ND*DD) return;
  int n = tid / DD, o = tid % DD;
  const float* xr = x + (size_t)n*FN;
  const float* wr = w + (size_t)o*FN;
  float acc = b[o];
  for (int i=0;i<FN;++i) acc += xr[i]*wr[i];
  float r = fmaxf(acc, 0.0f);
  out[tid] = r; h[tid] = r;
}

// ---- 2) h1 = relu(edge_attr @ nn1_w.T + nn1_b)   [NE x 128] ----
__global__ void k_edge_mlp1(const float* __restrict__ ea, const float* __restrict__ w,
                            const float* __restrict__ b, float* __restrict__ h1){
  int tid = blockIdx.x*blockDim.x + threadIdx.x;
  if (tid >= NE*HIDC) return;
  int e = tid / HIDC, c = tid % HIDC;
  const float* er = ea + (size_t)e*FE;
  const float* wr = w + (size_t)c*FE;
  float acc = b[c];
  for (int i=0;i<FE;++i) acc += er[i]*wr[i];
  h1[tid] = fmaxf(acc, 0.0f);
}

// ---- 3) pack A (h1 -> f16, WMMA 16-bit A 16x32 layout) ----
// Apack[((mtile*4+kstep)*32+lane)*8+v] = pack2(h1[m,k0], h1[m,k0+1])
__global__ void k_packA(const float* __restrict__ h1, uint32_t* __restrict__ ap){
  int idx = blockIdx.x*blockDim.x + threadIdx.x;
  if (idx >= MT*4*32*8) return;
  int v     = idx & 7;
  int lane  = (idx >> 3) & 31;
  int kstep = (idx >> 8) & 3;
  int mtile = idx >> 10;
  int m    = mtile*16 + (lane & 15);
  int half = lane >> 4;
  int kb = kstep*32 + ((v < 4) ? (2*v + half*8) : (16 + 2*(v-4) + half*8));
  union { uint32_t u; _Float16 hh[2]; } pk;
  pk.hh[0] = (_Float16)h1[(size_t)m*HIDC + kb];
  pk.hh[1] = (_Float16)h1[(size_t)m*HIDC + kb + 1];
  ap[idx] = pk.u;
}

// ---- 4) pack B (nn2_w.T -> f16, WMMA 16-bit B 32x16 layout) ----
__global__ void k_packB(const float* __restrict__ w2, uint32_t* __restrict__ bp){
  int idx = blockIdx.x*blockDim.x + threadIdx.x;
  if (idx >= NT*4*32*8) return;
  int v     = idx & 7;
  int lane  = (idx >> 3) & 31;
  int kstep = (idx >> 8) & 3;
  int ntile = idx >> 10;
  int n  = ntile*16 + (lane & 15);
  int kb = kstep*32 + ((v < 4) ? (2*v + (lane >> 4)*8) : (16 + 2*(v-4) + (lane >> 4)*8));
  union { uint32_t u; _Float16 hh[2]; } pk;
  pk.hh[0] = (_Float16)w2[(size_t)n*HIDC + kb];
  pk.hh[1] = (_Float16)w2[(size_t)n*HIDC + kb + 1];
  bp[idx] = pk.u;
}

// ---- 5) We = h1 @ nn2_w.T + nn2_b  (M=64000,N=4096,K=128) -> f16 ----
// Register-blocked 2M x 4N per wave (32 v_wmma per wave).
// A (shared by all 8 waves of the block) is staged in LDS with
// global_load_async_to_lds_b128 (ASYNCcnt path); B read direct from L2.
__global__ void __launch_bounds__(256)
k_gemm_we(const uint32_t* __restrict__ ap, const uint32_t* __restrict__ bp,
          const float* __restrict__ b2, _Float16* __restrict__ We){
  __shared__ __align__(32) uint32_t sA[2048];   // 2 M-tiles * 1024 dwords = 8KB
  int wave  = threadIdx.x >> 5;
  int lane  = threadIdx.x & 31;
  int mt0 = blockIdx.x * 2;                 // two consecutive M tiles
  int nt0 = (blockIdx.y*8 + wave) * 4;      // four consecutive N tiles

  // async-stage 8KB of packed A for this M-pair into LDS
  {
    const uint32_t* gsrc = ap + (size_t)mt0*1024 + threadIdx.x*4;
    uint32_t dst = (uint32_t)(uintptr_t)(&sA[threadIdx.x*4]);
    asm volatile("global_load_async_to_lds_b128 %0, %1, off"
                 :: "v"(dst), "v"((uint64_t)(uintptr_t)gsrc) : "memory");
    asm volatile("global_load_async_to_lds_b128 %0, %1, off"
                 :: "v"(dst + 4096u), "v"((uint64_t)(uintptr_t)(gsrc + 1024)) : "memory");
    asm volatile("s_wait_asynccnt 0x0" ::: "memory");
  }
  __syncthreads();

  v8f acc[2][4];
  #pragma unroll
  for (int i=0;i<2;++i)
    #pragma unroll
    for (int j=0;j<4;++j) acc[i][j] = (v8f){0.f,0.f,0.f,0.f,0.f,0.f,0.f,0.f};

  const uint32_t* bb = bp + (size_t)nt0*1024 + lane*8;
  #pragma unroll
  for (int ks = 0; ks < 4; ++ks){
    v16h A0 = *reinterpret_cast<const v16h*>(&sA[        ks*256 + lane*8]);
    v16h A1 = *reinterpret_cast<const v16h*>(&sA[1024 +  ks*256 + lane*8]);
    #pragma unroll
    for (int j = 0; j < 4; ++j){
      v16h B = *reinterpret_cast<const v16h*>(bb + (size_t)j*1024 + ks*256);
      acc[0][j] = __builtin_amdgcn_wmma_f32_16x16x32_f16(false, A0, false, B, (short)0, acc[0][j], false, false);
      acc[1][j] = __builtin_amdgcn_wmma_f32_16x16x32_f16(false, A1, false, B, (short)0, acc[1][j], false, false);
    }
  }

  int row0 = (lane < 16) ? 0 : 8;
  #pragma unroll
  for (int i=0;i<2;++i){
    #pragma unroll
    for (int j=0;j<4;++j){
      int col  = (nt0+j)*16 + (lane & 15);
      float bias = b2[col];
      #pragma unroll
      for (int v = 0; v < 8; ++v){
        int e = (mt0+i)*16 + row0 + v;
        We[(size_t)e*WEN + col] = (_Float16)(acc[i][j][v] + bias);
      }
    }
  }
}

// ---- 6) degree ----
__global__ void k_deg(const int* __restrict__ ei, float* __restrict__ deg){
  int e = blockIdx.x*blockDim.x + threadIdx.x;
  if (e >= NE) return;
  atomicAdd(&deg[ei[NE + e]], 1.0f);
}

// ---- 7) msg_e = out[src] @ We_e ; scatter-add into agg[dst] ----
// one wave per edge; lane handles output cols (2*lane, 2*lane+1); per-iteration
// the wave reads one full 128B We row (perfectly coalesced stream of fp16 We).
__global__ void k_msg(const _Float16* __restrict__ We, const float* __restrict__ out,
                      const int* __restrict__ ei, float* __restrict__ agg){
  int e    = blockIdx.x*8 + (threadIdx.x >> 5);
  int lane = threadIdx.x & 31;
  int src = ei[e], dst = ei[NE + e];
  const uint32_t* wrow = reinterpret_cast<const uint32_t*>(We) + (size_t)e*2048 + lane;
  const float* xr = out + (size_t)src*DD;
  float a0 = 0.f, a1 = 0.f;
  for (int i = 0; i < DD; ++i){
    __builtin_prefetch(wrow + (i+8)*32, 0, 0);
    float xi = xr[i];
    union { uint32_t u; _Float16 hh[2]; } pk; pk.u = wrow[i*32];
    a0 += xi * (float)pk.hh[0];
    a1 += xi * (float)pk.hh[1];
  }
  atomicAdd(&agg[(size_t)dst*DD + 2*lane    ], a0);
  atomicAdd(&agg[(size_t)dst*DD + 2*lane + 1], a1);
}

// ---- 8) m = relu(agg/deg + out @ root_w.T + conv_b) ----
__global__ void k_conv_update(const float* __restrict__ agg, const float* __restrict__ deg,
                              const float* __restrict__ out, const float* __restrict__ rw,
                              const float* __restrict__ cb, float* __restrict__ m){
  int tid = blockIdx.x*blockDim.x + threadIdx.x;
  if (tid >= ND*DD) return;
  int n = tid / DD, o = tid % DD;
  const float* xr = out + (size_t)n*DD;
  const float* wr = rw + (size_t)o*DD;
  float acc = cb[o];
  for (int i=0;i<DD;++i) acc += xr[i]*wr[i];
  acc += agg[tid] / fmaxf(deg[n], 1.0f);
  m[tid] = fmaxf(acc, 0.0f);
}

// ---- 9) GRU step (torch gate order r,z,n); writes h_new to hn ----
__global__ void k_gru(const float* __restrict__ m, const float* __restrict__ h,
                      const float* __restrict__ wih, const float* __restrict__ whh,
                      const float* __restrict__ bih, const float* __restrict__ bhh,
                      float* __restrict__ hn){
  int tid = blockIdx.x*blockDim.x + threadIdx.x;
  if (tid >= ND*DD) return;
  int n = tid / DD, j = tid % DD;
  const float* mr = m + (size_t)n*DD;
  const float* hr = h + (size_t)n*DD;
  float gir=0.f, giz=0.f, gin=0.f, ghr=0.f, ghz=0.f, ghn=0.f;
  for (int i=0;i<DD;++i){
    float mi = mr[i], hi = hr[i];
    gir += mi*wih[(size_t)j*DD + i];
    giz += mi*wih[(size_t)(DD+j)*DD + i];
    gin += mi*wih[(size_t)(2*DD+j)*DD + i];
    ghr += hi*whh[(size_t)j*DD + i];
    ghz += hi*whh[(size_t)(DD+j)*DD + i];
    ghn += hi*whh[(size_t)(2*DD+j)*DD + i];
  }
  float r  = sigf(gir + bih[j]      + ghr + bhh[j]);
  float z  = sigf(giz + bih[DD+j]   + ghz + bhh[DD+j]);
  float nn = tanhf(gin + bih[2*DD+j] + r*(ghn + bhh[2*DD+j]));
  hn[tid] = (1.f - z)*nn + z*hr[j];
}

__global__ void k_copy2(const float* __restrict__ hn, float* __restrict__ h,
                        float* __restrict__ out){
  int tid = blockIdx.x*blockDim.x + threadIdx.x;
  if (tid >= ND*DD) return;
  h[tid] = hn[tid]; out[tid] = hn[tid];
}

// ---- 10) Set2Set LSTM cell (one block = one segment, 64 threads) ----
__global__ void k_lstm(const float* __restrict__ qstar, float* __restrict__ hl,
                       float* __restrict__ cl, const float* __restrict__ wih,
                       const float* __restrict__ whh, const float* __restrict__ bih,
                       const float* __restrict__ bhh){
  __shared__ float sq[2*DD];
  __shared__ float sh[DD];
  int seg = blockIdx.x, d = threadIdx.x;
  sq[d]      = qstar[(size_t)seg*2*DD + d];
  sq[d+DD]   = qstar[(size_t)seg*2*DD + d + DD];
  sh[d]      = hl[(size_t)seg*DD + d];
  __syncthreads();
  float g[4];
  #pragma unroll
  for (int gate=0; gate<4; ++gate){
    int row = gate*DD + d;
    float acc = bih[row] + bhh[row];
    for (int i=0;i<2*DD;++i) acc += sq[i]*wih[(size_t)row*2*DD + i];
    for (int i=0;i<DD;++i)   acc += sh[i]*whh[(size_t)row*DD + i];
    g[gate] = acc;
  }
  float iv = sigf(g[0]), fv = sigf(g[1]), gv = tanhf(g[2]), ov = sigf(g[3]);
  float c = fv*cl[(size_t)seg*DD + d] + iv*gv;
  cl[(size_t)seg*DD + d] = c;
  hl[(size_t)seg*DD + d] = ov*tanhf(c);
}

// ---- 11) attention energies e_n = <out[n], hl[batch[n]]> ----
__global__ void k_attn_e(const float* __restrict__ out, const float* __restrict__ hl,
                         const int* __restrict__ batch, float* __restrict__ eo){
  int n = blockIdx.x*blockDim.x + threadIdx.x;
  if (n >= ND) return;
  const float* xr = out + (size_t)n*DD;
  const float* qr = hl + (size_t)batch[n]*DD;
  float acc = 0.f;
  for (int i=0;i<DD;++i) acc += xr[i]*qr[i];
  eo[n] = acc;
}

__global__ void k_init_seg(unsigned* __restrict__ mmax, float* __restrict__ denom){
  int s = threadIdx.x;
  if (s >= NSEGC) return;
  mmax[s] = enc_f(-FLT_MAX);
  denom[s] = 0.f;
}

__global__ void k_segmax(const float* __restrict__ eo, const int* __restrict__ batch,
                         unsigned* __restrict__ mmax){
  int n = blockIdx.x*blockDim.x + threadIdx.x;
  if (n >= ND) return;
  atomicMax(&mmax[batch[n]], enc_f(eo[n]));
}

__global__ void k_expdenom(const float* __restrict__ eo, const int* __restrict__ batch,
                           const unsigned* __restrict__ mmax, float* __restrict__ a,
                           float* __restrict__ denom){
  int n = blockIdx.x*blockDim.x + threadIdx.x;
  if (n >= ND) return;
  int b = batch[n];
  float av = expf(eo[n] - dec_f(mmax[b]));
  a[n] = av;
  atomicAdd(&denom[b], av);
}

__global__ void k_qstar_init(const float* __restrict__ hl, float* __restrict__ qstar){
  int tid = blockIdx.x*blockDim.x + threadIdx.x;
  if (tid >= NSEGC*2*DD) return;
  int seg = tid / (2*DD), c = tid % (2*DD);
  qstar[tid] = (c < DD) ? hl[(size_t)seg*DD + c] : 0.f;
}

__global__ void k_scatter_r(const float* __restrict__ out, const float* __restrict__ a,
                            const float* __restrict__ denom, const int* __restrict__ batch,
                            float* __restrict__ qstar){
  int tid = blockIdx.x*blockDim.x + threadIdx.x;
  if (tid >= ND*DD) return;
  int n = tid / DD, d = tid % DD;
  int b = batch[n];
  float w = a[n] / denom[b];
  atomicAdd(&qstar[(size_t)b*2*DD + DD + d], w*out[tid]);
}

// ---- 12) y = relu(qstar @ lin1_w.T + lin1_b) @ lin2_w.T + lin2_b ----
__global__ void k_readout(const float* __restrict__ qstar, const float* __restrict__ w1,
                          const float* __restrict__ b1, const float* __restrict__ w2,
                          const float* __restrict__ b2, float* __restrict__ y){
  __shared__ float red[DD];
  int seg = blockIdx.x, d = threadIdx.x;
  const float* q = qstar + (size_t)seg*2*DD;
  float acc = b1[d];
  for (int i=0;i<2*DD;++i) acc += q[i]*w1[(size_t)d*2*DD + i];
  red[d] = fmaxf(acc, 0.f)*w2[d];
  __syncthreads();
  for (int s = 32; s > 0; s >>= 1){
    if (d < s) red[d] += red[d+s];
    __syncthreads();
  }
  if (d == 0) y[seg] = red[0] + b2[0];
}

extern "C" void kernel_launch(void* const* d_in, const int* in_sizes, int n_in,
                              void* d_out, int out_size, void* d_ws, size_t ws_size,
                              hipStream_t stream) {
  const float* x        = (const float*)d_in[0];
  const float* edge_attr= (const float*)d_in[1];
  const float* lin0_w   = (const float*)d_in[2];
  const float* lin0_b   = (const float*)d_in[3];
  const float* nn1_w    = (const float*)d_in[4];
  const float* nn1_b    = (const float*)d_in[5];
  const float* nn2_w    = (const float*)d_in[6];
  const float* nn2_b    = (const float*)d_in[7];
  const float* root_w   = (const float*)d_in[8];
  const float* conv_b   = (const float*)d_in[9];
  const float* gru_w_ih = (const float*)d_in[10];
  const float* gru_w_hh = (const float*)d_in[11];
  const float* gru_b_ih = (const float*)d_in[12];
  const float* gru_b_hh = (const float*)d_in[13];
  const float* lstm_w_ih= (const float*)d_in[14];
  const float* lstm_w_hh= (const float*)d_in[15];
  const float* lstm_b_ih= (const float*)d_in[16];
  const float* lstm_b_hh= (const float*)d_in[17];
  const float* lin1_w   = (const float*)d_in[18];
  const float* lin1_b   = (const float*)d_in[19];
  const float* lin2_w   = (const float*)d_in[20];
  const float* lin2_b   = (const float*)d_in[21];
  const int*   edge_idx = (const int*)d_in[22];
  const int*   batch    = (const int*)d_in[23];
  float* y = (float*)d_out;

  // workspace carve-up (256B aligned)
  char* p = (char*)d_ws;
  auto carve = [&](size_t bytes) -> void* {
    void* r = (void*)p;
    p += (bytes + 255) & ~(size_t)255;
    return r;
  };
  float*    out   = (float*)   carve((size_t)ND*DD*4);
  float*    hbuf  = (float*)   carve((size_t)ND*DD*4);
  float*    mbuf  = (float*)   carve((size_t)ND*DD*4);
  float*    agg   = (float*)   carve((size_t)ND*DD*4);   // reused as GRU h_new
  float*    deg   = (float*)   carve((size_t)ND*4);
  float*    h1    = (float*)   carve((size_t)NE*HIDC*4);
  uint32_t* Apack = (uint32_t*)carve((size_t)MT*1024*4);
  uint32_t* Bpack = (uint32_t*)carve((size_t)NT*1024*4);
  _Float16* We16  = (_Float16*)carve((size_t)NE*WEN*2);
  float*    eo    = (float*)   carve((size_t)ND*4);
  float*    aat   = (float*)   carve((size_t)ND*4);
  unsigned* mmax  = (unsigned*)carve((size_t)NSEGC*4);
  float*    denom = (float*)   carve((size_t)NSEGC*4);
  float*    qstar = (float*)   carve((size_t)NSEGC*2*DD*4);
  float*    hl    = (float*)   carve((size_t)NSEGC*DD*4);
  float*    cl    = (float*)   carve((size_t)NSEGC*DD*4);

  const int B = 256;

  // front end
  k_lin0<<<(ND*DD + B-1)/B, B, 0, stream>>>(x, lin0_w, lin0_b, out, hbuf);
  k_edge_mlp1<<<(NE*HIDC + B-1)/B, B, 0, stream>>>(edge_attr, nn1_w, nn1_b, h1);
  k_packA<<<(MT*1024 + B-1)/B, B, 0, stream>>>(h1, Apack);
  k_packB<<<(NT*1024 + B-1)/B, B, 0, stream>>>(nn2_w, Bpack);
  k_gemm_we<<<dim3(MT/2, 8), B, 0, stream>>>(Apack, Bpack, nn2_b, We16);

  hipMemsetAsync(deg, 0, (size_t)ND*4, stream);
  k_deg<<<(NE + B-1)/B, B, 0, stream>>>(edge_idx, deg);

  // NCONV = 2 message-passing iterations
  for (int t = 0; t < 2; ++t){
    hipMemsetAsync(agg, 0, (size_t)ND*DD*4, stream);
    k_msg<<<NE/8, B, 0, stream>>>(We16, out, edge_idx, agg);
    k_conv_update<<<(ND*DD + B-1)/B, B, 0, stream>>>(agg, deg, out, root_w, conv_b, mbuf);
    k_gru<<<(ND*DD + B-1)/B, B, 0, stream>>>(mbuf, hbuf, gru_w_ih, gru_w_hh,
                                             gru_b_ih, gru_b_hh, agg);
    k_copy2<<<(ND*DD + B-1)/B, B, 0, stream>>>(agg, hbuf, out);
  }

  // Set2Set (STEPS = 3)
  hipMemsetAsync(qstar, 0, (size_t)NSEGC*2*DD*4, stream);
  hipMemsetAsync(hl, 0, (size_t)NSEGC*DD*4, stream);
  hipMemsetAsync(cl, 0, (size_t)NSEGC*DD*4, stream);
  for (int s = 0; s < 3; ++s){
    k_lstm<<<NSEGC, DD, 0, stream>>>(qstar, hl, cl, lstm_w_ih, lstm_w_hh,
                                     lstm_b_ih, lstm_b_hh);
    k_attn_e<<<(ND + B-1)/B, B, 0, stream>>>(out, hl, batch, eo);
    k_init_seg<<<1, NSEGC, 0, stream>>>(mmax, denom);
    k_segmax<<<(ND + B-1)/B, B, 0, stream>>>(eo, batch, mmax);
    k_expdenom<<<(ND + B-1)/B, B, 0, stream>>>(eo, batch, mmax, aat, denom);
    k_qstar_init<<<(NSEGC*2*DD + B-1)/B, B, 0, stream>>>(hl, qstar);
    k_scatter_r<<<(ND*DD + B-1)/B, B, 0, stream>>>(out, aat, denom, batch, qstar);
  }

  k_readout<<<NSEGC, DD, 0, stream>>>(qstar, lin1_w, lin1_b, lin2_w, lin2_b, y);
}
